// MARN_89232240541884
// MI455X (gfx1250) — matrix operand order repaired and checked by reading, not compiled, not run on hardware
//
#include <hip/hip_runtime.h>

// ---------------------------------------------------------------------------
// MARN recurrent net, single persistent kernel per call.
// Every per-step stage is independent across batch rows; the recurrence
// couples time steps only per-row. Each workgroup owns a 16-row batch stripe
// and runs all T=256 steps with state in registers (c) and LDS (h, z,
// intermediates). bf16 WMMA (v_wmma_f32_16x16x32_bf16), f32 accumulate.
// Weights pre-transposed to bf16 [N,K] once; resident in 192MB L2.
// ---------------------------------------------------------------------------

#define T_STEPS 256
#define BATCH   512
#define CDIM    128

typedef __attribute__((ext_vector_type(16))) __bf16          v16bf;
typedef __attribute__((ext_vector_type(2)))  __bf16          v2bf;
typedef __attribute__((ext_vector_type(8)))  float           v8f;
typedef __attribute__((ext_vector_type(4)))  float           v4f;
typedef __attribute__((ext_vector_type(8)))  unsigned short  v8u;
typedef __attribute__((ext_vector_type(16))) unsigned short  v16u;
typedef __attribute__((ext_vector_type(4)))  unsigned        v4u;

__device__ __forceinline__ unsigned short f2bf(float f) {
    unsigned u = __builtin_bit_cast(unsigned, f);
    u += 0x7fffu + ((u >> 16) & 1u);          // round-to-nearest-even
    return (unsigned short)(u >> 16);
}

// packed f32x2 -> bf16x2 (low = a, high = b)
__device__ __forceinline__ unsigned f2bf_pk(float a, float b) {
#if __has_builtin(__builtin_amdgcn_cvt_pk_bf16_f32)
    v2bf r = __builtin_amdgcn_cvt_pk_bf16_f32(a, b);
    return __builtin_bit_cast(unsigned, r);
#else
    return (unsigned)f2bf(a) | ((unsigned)f2bf(b) << 16);
#endif
}

__device__ __forceinline__ v16bf mk16(v8u lo, v8u hi) {
    v16u v = __builtin_shufflevector(lo, hi, 0,1,2,3,4,5,6,7,8,9,10,11,12,13,14,15);
    return __builtin_bit_cast(v16bf, v);
}

// A-fragment (16x32 bf16): row-major [16,K] tile (row0 always 0 in-stripe).
__device__ __forceinline__ v16bf load_a_bf(const unsigned short* A, int lda,
                                           int kb, int lane) {
    int r = lane & 15, h = lane >> 4;
    const unsigned short* p = A + (r * lda + kb + 8 * h);
    v8u lo = *(const v8u*)p;
    v8u hi = *(const v8u*)(p + 16);
    return mk16(lo, hi);
}

// B-fragment (32x16 bf16) from pre-transposed weights BT[N,K] (ldb=K).
__device__ __forceinline__ v16bf load_b(const unsigned short* BT, int ldb,
                                        int col0, int kb, int lane) {
    int n = lane & 15, kh = lane >> 4;
    const unsigned short* p = BT + (size_t)(col0 + n) * ldb + kb + 8 * kh;
    v8u lo = *(const v8u*)p;
    v8u hi = *(const v8u*)(p + 16);
    return mk16(lo, hi);
}

__device__ __forceinline__ v8f wmma_bf(v16bf a, v16bf b, v8f c) {
    return __builtin_amdgcn_wmma_f32_16x16x32_bf16(false, a, false, b,
                                                   (short)0, c, false, false);
}

__device__ __forceinline__ float fast_sigmoid(float x) {
    return __builtin_amdgcn_rcpf(1.0f + __expf(-x));
}
__device__ __forceinline__ float fast_tanh(float x) {
    return 1.0f - 2.0f * __builtin_amdgcn_rcpf(1.0f + __expf(2.0f * x));
}

// one-time prep: weights fp32 [K,N] -> bf16 [N,K]
__global__ void transpose_to_bf16(const float* __restrict__ src,
                                  unsigned short* __restrict__ dst,
                                  int K, int N) {
    int idx = blockIdx.x * blockDim.x + threadIdx.x;
    if (idx >= K * N) return;
    int k = idx / N, n = idx % N;
    dst[n * K + k] = f2bf(src[idx]);
}

struct Params {
    const float* eeg; const float* eog;
    const float *Wb, *Ub, *Vb;
    const unsigned short *WwT, *UwT, *VwT, *A1T, *A2T, *D10T, *D20T, *D11T, *D21T;
    const float *a1, *a2, *e10, *e20, *e11, *e21;
    float* out;
};

// LDS layout (bytes):
//   hf   [2][16][128]  f32   @      0  (16384)  h fp32 (attention scaling)
//   lg   [16][1024]    f32   @  16384  (65536)  attention logits
//   xs   [2][16][128]  bf16  @  81920  ( 8192)  x tile, converted
//   hcat [2][16][256]  bf16  @  90112  (16384)  concat(h0,h1), double-buffered
//   zb   [2][16][128]  bf16  @ 106496  ( 8192)  z (recurrent)
//   t1s  [16][128]     bf16  @ 114688  ( 4096)  attn hidden
//   atts [16][1024]    bf16  @ 118784  (32768)  softmax*tile(h)
//   sb   [2][16][128]  bf16  @ 151552  ( 8192)  dim-reduce hidden
#define SMEM_BYTES 159744

__global__ __launch_bounds__(512) void marn_kernel(Params p) {
    extern __shared__ char smem[];
    float*          hf   = (float*)(smem);
    float*          lg   = (float*)(smem + 16384);
    unsigned short* xs   = (unsigned short*)(smem + 81920);
    unsigned short* hcat = (unsigned short*)(smem + 90112);
    unsigned short* zb   = (unsigned short*)(smem + 106496);
    unsigned short* t1s  = (unsigned short*)(smem + 114688);
    unsigned short* atts = (unsigned short*)(smem + 118784);
    unsigned short* sb   = (unsigned short*)(smem + 151552);

    const int tid   = threadIdx.x;
    const int lane  = tid & 31;
    const int w     = tid >> 5;            // wave 0..15
    const int row0g = blockIdx.x * 16;     // batch stripe
    const int h2    = lane >> 4;

    // fixed per-wave roles (constant across all t)
    const int Lm   = w >> 3;               // modality for LSTHM/D1/D2 phases
    const int Lc   = (w & 7) * 16;         // 16-col tile within C
    const int ccol = Lc + (lane & 15);

    // hoisted time-invariant bias loads
    float wb[4], ubvb[4];
#pragma unroll
    for (int g = 0; g < 4; ++g) {
        wb[g]   = p.Wb[g * CDIM + ccol];
        ubvb[g] = p.Ub[g * CDIM + ccol] + p.Vb[g * CDIM + ccol];
    }
    const float b_a1 = p.a1[(w & 7) * 16 + (lane & 15)];   // used by waves 0..7
    float b_a2[4];
#pragma unroll
    for (int q = 0; q < 4; ++q) b_a2[q] = p.a2[w * 64 + q * 16 + (lane & 15)];
    const float b_e1 = (Lm ? p.e11 : p.e10)[ccol];
    const float b_e2 = (Lm ? p.e21 : p.e20)[ccol];
    const unsigned short* D1T = Lm ? p.D11T : p.D10T;
    const unsigned short* D2T = Lm ? p.D21T : p.D20T;

    float creg[8];                         // LSTM cell state lives in VGPRs

    // x-convert role
    const int xm   = tid >> 8;             // modality
    const int xr   = (tid >> 4) & 15;      // row in stripe
    const int xseg = (tid & 15) * 8;       // 8-elem segment

    for (int t = 0; t < T_STEPS; ++t) {
        const bool first = (t == 0);
        unsigned short* hc_w = hcat + (t & 1) * 4096;        // write buffer
        const unsigned short* hc_r = hcat + ((t + 1) & 1) * 4096;  // prev h

        // ---- phase X: stream x tile -> bf16 LDS; prefetch next step ----
        {
            const float* xp = (xm ? p.eog : p.eeg)
                            + (size_t)t * (BATCH * CDIM)
                            + (size_t)(row0g + xr) * CDIM + xseg;
            v4f x0 = *(const v4f*)xp;
            v4f x1 = *(const v4f*)(xp + 4);
            v4u pk;
            pk[0] = f2bf_pk(x0[0], x0[1]);
            pk[1] = f2bf_pk(x0[2], x0[3]);
            pk[2] = f2bf_pk(x1[0], x1[1]);
            pk[3] = f2bf_pk(x1[2], x1[3]);
            *(v4u*)(xs + (xm * 16 + xr) * CDIM + xseg) = pk;
            if (t + 1 < T_STEPS) __builtin_prefetch(xp + BATCH * CDIM, 0, 0);
        }
        __syncthreads();

        // ---- phase L: LSTHM, all 4 gate column-blocks share each A-frag ----
        {
            v8f g0 = {}, g1 = {}, g2 = {}, g3 = {};
            const unsigned short* xA = xs + Lm * 16 * CDIM;
#pragma unroll
            for (int kb = 0; kb < CDIM; kb += 32) {
                v16bf a = load_a_bf(xA, CDIM, kb, lane);
                g0 = wmma_bf(a, load_b(p.WwT, CDIM, 0 * CDIM + Lc, kb, lane), g0);
                g1 = wmma_bf(a, load_b(p.WwT, CDIM, 1 * CDIM + Lc, kb, lane), g1);
                g2 = wmma_bf(a, load_b(p.WwT, CDIM, 2 * CDIM + Lc, kb, lane), g2);
                g3 = wmma_bf(a, load_b(p.WwT, CDIM, 3 * CDIM + Lc, kb, lane), g3);
            }
            if (!first) {
                const unsigned short* hA = hc_r + Lm * CDIM;    // lda = 2C
#pragma unroll
                for (int kb = 0; kb < CDIM; kb += 32) {
                    v16bf a = load_a_bf(hA, 2 * CDIM, kb, lane);
                    g0 = wmma_bf(a, load_b(p.UwT, CDIM, 0 * CDIM + Lc, kb, lane), g0);
                    g1 = wmma_bf(a, load_b(p.UwT, CDIM, 1 * CDIM + Lc, kb, lane), g1);
                    g2 = wmma_bf(a, load_b(p.UwT, CDIM, 2 * CDIM + Lc, kb, lane), g2);
                    g3 = wmma_bf(a, load_b(p.UwT, CDIM, 3 * CDIM + Lc, kb, lane), g3);
                }
                const unsigned short* zA = zb + Lm * 16 * CDIM;
#pragma unroll
                for (int kb = 0; kb < CDIM; kb += 32) {
                    v16bf a = load_a_bf(zA, CDIM, kb, lane);
                    g0 = wmma_bf(a, load_b(p.VwT, CDIM, 0 * CDIM + Lc, kb, lane), g0);
                    g1 = wmma_bf(a, load_b(p.VwT, CDIM, 1 * CDIM + Lc, kb, lane), g1);
                    g2 = wmma_bf(a, load_b(p.VwT, CDIM, 2 * CDIM + Lc, kb, lane), g2);
                    g3 = wmma_bf(a, load_b(p.VwT, CDIM, 3 * CDIM + Lc, kb, lane), g3);
                }
            }
            const float gb0 = wb[0] + (first ? 0.f : ubvb[0]);
            const float gb1 = wb[1] + (first ? 0.f : ubvb[1]);
            const float gb2 = wb[2] + (first ? 0.f : ubvb[2]);
            const float gb3 = wb[3] + (first ? 0.f : ubvb[3]);
            float hv[8];
#pragma unroll
            for (int i = 0; i < 8; ++i) {
                float f  = fast_sigmoid(g0[i] + gb0);
                float ii = fast_sigmoid(g1[i] + gb1);
                float o  = fast_sigmoid(g2[i] + gb2);
                float ch = fast_tanh   (g3[i] + gb3);
                float cp = first ? 0.f : creg[i];
                float cn = f * cp + ii * ch;
                creg[i]  = cn;
                hv[i] = fast_tanh(cn) * o;
                hf[Lm * 2048 + (i + 8 * h2) * CDIM + ccol] = hv[i];
            }
            // write new h into the other hcat buffer: no read hazard, no barrier
#pragma unroll
            for (int i = 0; i < 8; i += 2) {
                unsigned pk = f2bf_pk(hv[i], hv[i + 1]);
                hc_w[(i     + 8 * h2) * (2 * CDIM) + Lm * CDIM + ccol] = (unsigned short)pk;
                hc_w[(i + 1 + 8 * h2) * (2 * CDIM) + Lm * CDIM + ccol] = (unsigned short)(pk >> 16);
            }
        }
        __syncthreads();

        // ---- phase A1: t1 = tanh(hcat @ A1 + a1), waves 0..7 ----
        if (w < 8) {
            v8f acc = {};
            const int col0 = (w & 7) * 16;
#pragma unroll
            for (int kb = 0; kb < 2 * CDIM; kb += 32)
                acc = wmma_bf(load_a_bf(hc_w, 2 * CDIM, kb, lane),
                              load_b(p.A1T, 2 * CDIM, col0, kb, lane), acc);
#pragma unroll
            for (int i = 0; i < 8; i += 2) {
                unsigned pk = f2bf_pk(fast_tanh(acc[i]     + b_a1),
                                      fast_tanh(acc[i + 1] + b_a1));
                t1s[(i     + 8 * h2) * CDIM + col0 + (lane & 15)] = (unsigned short)pk;
                t1s[(i + 1 + 8 * h2) * CDIM + col0 + (lane & 15)] = (unsigned short)(pk >> 16);
            }
        }
        __syncthreads();

        // ---- phase A2: logits = t1 @ A2 + a2 (4 col tiles per wave) ----
        {
            v8f q0 = {}, q1 = {}, q2 = {}, q3 = {};
#pragma unroll
            for (int kb = 0; kb < CDIM; kb += 32) {
                v16bf a = load_a_bf(t1s, CDIM, kb, lane);
                q0 = wmma_bf(a, load_b(p.A2T, CDIM, w * 64 +  0, kb, lane), q0);
                q1 = wmma_bf(a, load_b(p.A2T, CDIM, w * 64 + 16, kb, lane), q1);
                q2 = wmma_bf(a, load_b(p.A2T, CDIM, w * 64 + 32, kb, lane), q2);
                q3 = wmma_bf(a, load_b(p.A2T, CDIM, w * 64 + 48, kb, lane), q3);
            }
#pragma unroll
            for (int i = 0; i < 8; ++i) {
                int gr = (i + 8 * h2) * 1024 + w * 64 + (lane & 15);
                lg[gr +  0] = q0[i] + b_a2[0];
                lg[gr + 16] = q1[i] + b_a2[1];
                lg[gr + 32] = q2[i] + b_a2[2];
                lg[gr + 48] = q3[i] + b_a2[3];
            }
        }
        __syncthreads();

        // ---- phase SM: softmax over NA=4 + scale by tiled h (vectorized) ----
        {
            int base = tid * 8, r = base >> 8, j0 = base & 255;
            const float* lrow = lg + r * 1024;
            v8f l0 = *(const v8f*)(lrow + j0);
            v8f l1 = *(const v8f*)(lrow + 256 + j0);
            v8f l2 = *(const v8f*)(lrow + 512 + j0);
            v8f l3 = *(const v8f*)(lrow + 768 + j0);
            v8f m0 = *(const v8f*)(hf + r * CDIM + (j0 & 127));
            v8f m1 = *(const v8f*)(hf + 2048 + r * CDIM + (j0 & 127));
            float o0[8], o1[8], o2[8], o3[8];
#pragma unroll
            for (int e = 0; e < 8; ++e) {
                float mx = fmaxf(fmaxf(l0[e], l1[e]), fmaxf(l2[e], l3[e]));
                float e0 = __expf(l0[e] - mx), e1 = __expf(l1[e] - mx);
                float e2 = __expf(l2[e] - mx), e3 = __expf(l3[e] - mx);
                float inv = __builtin_amdgcn_rcpf(e0 + e1 + e2 + e3);
                o0[e] = e0 * inv * m0[e];
                o1[e] = e1 * inv * m0[e];
                o2[e] = e2 * inv * m1[e];
                o3[e] = e3 * inv * m1[e];
            }
            unsigned short* arow = atts + r * 1024 + j0;
#pragma unroll
            for (int e = 0; e < 8; e += 2) {
                *(unsigned*)(arow +       e) = f2bf_pk(o0[e], o0[e + 1]);
                *(unsigned*)(arow + 256 + e) = f2bf_pk(o1[e], o1[e + 1]);
                *(unsigned*)(arow + 512 + e) = f2bf_pk(o2[e], o2[e + 1]);
                *(unsigned*)(arow + 768 + e) = f2bf_pk(o3[e], o3[e + 1]);
            }
        }
        __syncthreads();

        // ---- phase D1: s = tanh(att[:, m*512:+512] @ D1m + e1m) ----
        {
            v8f acc = {};
            const unsigned short* A = atts + Lm * 512;     // lda = 1024
#pragma unroll
            for (int kb = 0; kb < 512; kb += 32)
                acc = wmma_bf(load_a_bf(A, 1024, kb, lane),
                              load_b(D1T, 512, Lc, kb, lane), acc);
#pragma unroll
            for (int i = 0; i < 8; i += 2) {
                unsigned pk = f2bf_pk(fast_tanh(acc[i]     + b_e1),
                                      fast_tanh(acc[i + 1] + b_e1));
                sb[Lm * 2048 + (i     + 8 * h2) * CDIM + ccol] = (unsigned short)pk;
                sb[Lm * 2048 + (i + 1 + 8 * h2) * CDIM + ccol] = (unsigned short)(pk >> 16);
            }
        }
        __syncthreads();

        // ---- phase D2: z = s @ D2m + e2m -> d_out + recurrent z ----
        {
            v8f acc = {};
            const unsigned short* A = sb + Lm * 2048;
#pragma unroll
            for (int kb = 0; kb < CDIM; kb += 32)
                acc = wmma_bf(load_a_bf(A, CDIM, kb, lane),
                              load_b(D2T, CDIM, Lc, kb, lane), acc);
            float zv[8];
#pragma unroll
            for (int i = 0; i < 8; ++i) {
                zv[i] = acc[i] + b_e2;
                p.out[(size_t)t * (BATCH * 256) + (size_t)(row0g + i + 8 * h2) * 256
                      + Lm * CDIM + ccol] = zv[i];
            }
#pragma unroll
            for (int i = 0; i < 8; i += 2) {
                unsigned pk = f2bf_pk(zv[i], zv[i + 1]);
                zb[Lm * 2048 + (i     + 8 * h2) * CDIM + ccol] = (unsigned short)pk;
                zb[Lm * 2048 + (i + 1 + 8 * h2) * CDIM + ccol] = (unsigned short)(pk >> 16);
            }
        }
        // no barrier needed here: next phase-X barrier orders zb/sb reuse
    }
}

// ---------------------------------------------------------------------------
extern "C" void kernel_launch(void* const* d_in, const int* in_sizes, int n_in,
                              void* d_out, int out_size, void* d_ws, size_t ws_size,
                              hipStream_t stream) {
    const float* eeg = (const float*)d_in[0];
    const float* eog = (const float*)d_in[1];
    const float* Ww  = (const float*)d_in[2];   const float* Wb  = (const float*)d_in[3];
    const float* Uw  = (const float*)d_in[4];   const float* Ub  = (const float*)d_in[5];
    const float* Vw  = (const float*)d_in[6];   const float* Vb  = (const float*)d_in[7];
    const float* A1  = (const float*)d_in[8];   const float* a1  = (const float*)d_in[9];
    const float* A2  = (const float*)d_in[10];  const float* a2  = (const float*)d_in[11];
    const float* D10 = (const float*)d_in[12];  const float* e10 = (const float*)d_in[13];
    const float* D20 = (const float*)d_in[14];  const float* e20 = (const float*)d_in[15];
    const float* D11 = (const float*)d_in[16];  const float* e11 = (const float*)d_in[17];
    const float* D21 = (const float*)d_in[18];  const float* e21 = (const float*)d_in[19];

    // workspace: transposed bf16 weights only (1 MB)
    char* ws = (char*)d_ws;
    unsigned short* WwT  = (unsigned short*)(ws + 0);
    unsigned short* UwT  = (unsigned short*)(ws + 131072);
    unsigned short* VwT  = (unsigned short*)(ws + 262144);
    unsigned short* A1T  = (unsigned short*)(ws + 393216);
    unsigned short* A2T  = (unsigned short*)(ws + 458752);
    unsigned short* D10T = (unsigned short*)(ws + 720896);
    unsigned short* D20T = (unsigned short*)(ws + 851968);
    unsigned short* D11T = (unsigned short*)(ws + 884736);
    unsigned short* D21T = (unsigned short*)(ws + 1015808);

    auto prep = [&](const float* src, unsigned short* dst, int K, int N) {
        int n = K * N;
        hipLaunchKernelGGL(transpose_to_bf16, dim3((n + 255) / 256), dim3(256),
                           0, stream, src, dst, K, N);
    };
    prep(Ww,  WwT,  128, 512);
    prep(Uw,  UwT,  128, 512);
    prep(Vw,  VwT,  128, 512);
    prep(A1,  A1T,  256, 128);
    prep(A2,  A2T,  128, 1024);
    prep(D10, D10T, 512, 128);
    prep(D20, D20T, 128, 128);
    prep(D11, D11T, 512, 128);
    prep(D21, D21T, 128, 128);

    Params p;
    p.eeg = eeg; p.eog = eog; p.Wb = Wb; p.Ub = Ub; p.Vb = Vb;
    p.WwT = WwT; p.UwT = UwT; p.VwT = VwT; p.A1T = A1T; p.A2T = A2T;
    p.D10T = D10T; p.D20T = D20T; p.D11T = D11T; p.D21T = D21T;
    p.a1 = a1; p.a2 = a2; p.e10 = e10; p.e20 = e20; p.e11 = e11; p.e21 = e21;
    p.out = (float*)d_out;

    hipLaunchKernelGGL(marn_kernel, dim3(BATCH / 16), dim3(512), SMEM_BYTES,
                       stream, p);
    (void)in_sizes; (void)n_in; (void)out_size; (void)ws_size;
}
